// DenseFlashAttention_2465311228657
// MI455X (gfx1250) — compile-verified, matrix operand order checked by reading, not run on hardware
//
#include <hip/hip_runtime.h>
#include <hip/hip_bf16.h>

typedef __attribute__((ext_vector_type(16))) _Float16 v16h;
typedef __attribute__((ext_vector_type(2)))  _Float16 h2;
typedef __attribute__((ext_vector_type(8)))  float    v8f;

#define HEADS 4
#define FDIM  64
#define MIDD  32

// ---------------- helpers ----------------
__device__ __forceinline__ float softplusf(float x) {
    return (x > 20.f) ? x : log1pf(__expf(x));
}
__device__ __forceinline__ float n2nf(float v) {
    if (v != v) return 0.f;
    if (v >  3.402823466e38f) return  3.402823466e38f;
    if (v < -3.402823466e38f) return -3.402823466e38f;
    return v;
}
// float atomic max via sign-split int/uint atomics (init pattern 0xFF works)
__device__ __forceinline__ void atomicMaxFloat(float* addr, float v) {
    if (v >= 0.f) atomicMax((int*)addr, __float_as_int(v));
    else          atomicMin((unsigned int*)addr, __float_as_uint(v));
}

// ---------------- 0. weight transpose + f16 convert ----------------
// wT layout: [mat 0..11 = (We,Wr,Wt)xHEADS, 12 = Wout][col][k]  (column-major f16)
__global__ void prep_w_kernel(const float* __restrict__ We,
                              const float* __restrict__ Wr,
                              const float* __restrict__ Wt,
                              const float* __restrict__ Wout,
                              _Float16* __restrict__ wT) {
    int idx = blockIdx.x * blockDim.x + threadIdx.x;
    if (idx >= 13 * FDIM * FDIM) return;
    int mat = idx >> 12;              // /4096
    int rem = idx & 4095;
    int col = rem >> 6;
    int k   = rem & 63;
    float v;
    if (mat < 12) {
        int grp = mat >> 2;           // 0=We 1=Wr 2=Wt
        int h   = mat & 3;
        const float* W = (grp == 0 ? We : (grp == 1 ? Wr : Wt));
        v = W[((size_t)h * FDIM + k) * FDIM + col];
    } else {
        v = Wout[(size_t)k * FDIM + col];
    }
    wT[idx] = (_Float16)v;
}

// ---------------- 1. layernorm ----------------
__global__ void ln_kernel(const float* __restrict__ x,
                          const float* __restrict__ g,
                          const float* __restrict__ b,
                          float* __restrict__ xn,
                          _Float16* __restrict__ xnb, int N) {
    int warp = (blockIdx.x * blockDim.x + threadIdx.x) >> 5;
    int lane = threadIdx.x & 31;
    if (warp >= N) return;
    const float* row = x + (size_t)warp * FDIM;
    float v0 = row[lane], v1 = row[lane + 32];
    float s = v0 + v1;
    for (int m = 16; m >= 1; m >>= 1) s += __shfl_xor(s, m, 32);
    float mu = s * (1.f / 64.f);
    float c0 = v0 - mu, c1 = v1 - mu;
    float q = c0 * c0 + c1 * c1;
    for (int m = 16; m >= 1; m >>= 1) q += __shfl_xor(q, m, 32);
    float rinv = rsqrtf(q * (1.f / 64.f) + 1e-5f);
    float o0 = g[lane]      * c0 * rinv + b[lane];
    float o1 = g[lane + 32] * c1 * rinv + b[lane + 32];
    size_t base = (size_t)warp * FDIM;
    xn[base + lane]       = o0;
    xn[base + lane + 32]  = o1;
    xnb[base + lane]      = (_Float16)o0;
    xnb[base + lane + 32] = (_Float16)o1;
}

// ---------------- 2. e/r/t projections via WMMA ----------------
// one wave computes a 16x64 output tile for one (matrix, head)
__global__ void proj_kernel(const _Float16* __restrict__ xnb,
                            const _Float16* __restrict__ wT,
                            _Float16* __restrict__ et,
                            _Float16* __restrict__ rt,
                            _Float16* __restrict__ tt,
                            int N, int NT) {
    int wid  = (blockIdx.x * blockDim.x + threadIdx.x) >> 5;
    int lane = threadIdx.x & 31;
    int total = 3 * HEADS * NT;
    if (wid >= total) return;
    int mat  = wid / (HEADS * NT);
    int rem  = wid - mat * (HEADS * NT);
    int h    = rem / NT;
    int tile = rem - h * NT;

    const _Float16* Wb = wT + ((size_t)(mat * HEADS + h)) * FDIM * FDIM;
    _Float16* out = (mat == 0 ? et : (mat == 1 ? rt : tt)) + (size_t)h * N * FDIM;

    int lrow = lane & 15;
    int lhi  = (lane >> 4) & 1;
    int row  = tile * 16 + lrow;
    if (row >= N) row = N - 1;

    // A fragments (16-bit A 16x32 layout), two k-steps K=[0,32) and [32,64)
    const h2* xr = (const h2*)(xnb + (size_t)row * FDIM);
    v16h a0, a1;
#pragma unroll
    for (int i = 0; i < 8; ++i) {
        int k0 = ((i < 4) ? 0 : 16) + lhi * 8 + (i & 3) * 2;
        h2 p0 = xr[k0 >> 1];
        h2 p1 = xr[(32 + k0) >> 1];
        a0[2 * i] = p0[0]; a0[2 * i + 1] = p0[1];
        a1[2 * i] = p1[0]; a1[2 * i + 1] = p1[1];
    }
    bool full = (tile * 16 + 16 <= N);
#pragma unroll
    for (int nt = 0; nt < 4; ++nt) {
        int col = nt * 16 + lrow;
        const h2* wr = (const h2*)(Wb + (size_t)col * FDIM);
        v16h b0, b1;
#pragma unroll
        for (int i = 0; i < 8; ++i) {
            int k0 = ((i < 4) ? 0 : 16) + lhi * 8 + (i & 3) * 2;
            h2 p0 = wr[k0 >> 1];
            h2 p1 = wr[(32 + k0) >> 1];
            b0[2 * i] = p0[0]; b0[2 * i + 1] = p0[1];
            b1[2 * i] = p1[0]; b1[2 * i + 1] = p1[1];
        }
        v8f c = {};
        c = __builtin_amdgcn_wmma_f32_16x16x32_f16(false, a0, false, b0, (short)0, c, false, false);
        c = __builtin_amdgcn_wmma_f32_16x16x32_f16(false, a1, false, b1, (short)0, c, false, false);
        if (full) {
#pragma unroll
            for (int g2 = 0; g2 < 8; ++g2) {
                int r2 = tile * 16 + g2 + lhi * 8;
                out[(size_t)r2 * FDIM + col] = (_Float16)c[g2];
            }
        } else {
#pragma unroll
            for (int g2 = 0; g2 < 8; ++g2) {
                int r2 = tile * 16 + g2 + lhi * 8;
                if (r2 < N) out[(size_t)r2 * FDIM + col] = (_Float16)c[g2];
            }
        }
    }
}

// ---------------- 3. per-(head,node) scores + SiLU MLPs ----------------
__global__ void nodehead_kernel(const _Float16* __restrict__ et,
                                const float* __restrict__ rs, const float* __restrict__ ts,
                                const float* __restrict__ Wd1, const float* __restrict__ bd1,
                                const float* __restrict__ Wd2, const float* __restrict__ bd2,
                                const float* __restrict__ Wt1, const float* __restrict__ bt1,
                                const float* __restrict__ Wt2, const float* __restrict__ bt2,
                                float* __restrict__ se, float* __restrict__ st,
                                float* __restrict__ hdec, float* __restrict__ htmp, int N) {
    int t = blockIdx.x * blockDim.x + threadIdx.x;
    if (t >= HEADS * N) return;
    int h = t / N;
    int n = t - h * N;
    const _Float16* er = et + ((size_t)h * N + n) * FDIM;
    float ef[FDIM];
#pragma unroll
    for (int f = 0; f < FDIM; ++f) ef[f] = (float)er[f];

    float sacc = 0.f, tacc = 0.f;
#pragma unroll
    for (int f = 0; f < FDIM; ++f) {
        sacc += ef[f] * rs[h * FDIM + f];
        tacc += ef[f] * ts[h * FDIM + f];
    }
    se[t] = sacc;
    st[t] = tacc;

    float dacc = bd2[h], macc = bt2[h];
    for (int m = 0; m < MIDD; ++m) {
        float a1 = bd1[h * MIDD + m];
        float a2 = bt1[h * MIDD + m];
#pragma unroll
        for (int f = 0; f < FDIM; ++f) {
            a1 += ef[f] * Wd1[((size_t)h * FDIM + f) * MIDD + m];
            a2 += ef[f] * Wt1[((size_t)h * FDIM + f) * MIDD + m];
        }
        dacc += (a1 / (1.f + __expf(-a1))) * Wd2[h * MIDD + m];
        macc += (a2 / (1.f + __expf(-a2))) * Wt2[h * MIDD + m];
    }
    hdec[t] = dacc;
    htmp[t] = macc;
}

// ---------------- 4. edge logits + segment max ----------------
__global__ void edge_logits_kernel(const int* __restrict__ snd, const int* __restrict__ rcv,
                                   const float* __restrict__ elen,
                                   const float* __restrict__ se, const float* __restrict__ st,
                                   const float* __restrict__ hdec, const float* __restrict__ htmp,
                                   const float* __restrict__ log_scale,
                                   const float* __restrict__ temp_bias,
                                   const float* __restrict__ temp_weight,
                                   float* __restrict__ rlbuf, float* __restrict__ tlbuf,
                                   float* __restrict__ maxr, float* __restrict__ maxt,
                                   int N, int E) {
    int e = blockIdx.x * blockDim.x + threadIdx.x;
    if (e >= E) return;
    int s = snd[e], r = rcv[e];
    float len = elen[e];
#pragma unroll
    for (int h = 0; h < HEADS; ++h) {
        float scale = softplusf(log_scale[h]);
        float rl = se[h * N + s] - se[h * N + r] - (scale + hdec[h * N + r]) * len;
        float rt = softplusf(temp_bias[h] + temp_weight[h] * len + htmp[h * N + r]);
        rl = rl / (rt + 1e-4f);
        float tl = st[h * N + s] - st[h * N + r];
        rlbuf[(size_t)h * E + e] = rl;
        tlbuf[(size_t)h * E + e] = tl;
        atomicMaxFloat(&maxr[h * N + r], rl);
        atomicMaxFloat(&maxt[h * N + r], tl);
    }
}

// ---------------- 5. exp + segment sum ----------------
__global__ void edge_exp_kernel(const int* __restrict__ rcv,
                                float* __restrict__ rlbuf, float* __restrict__ tlbuf,
                                const float* __restrict__ maxr, const float* __restrict__ maxt,
                                float* __restrict__ denr, float* __restrict__ dent,
                                int N, int E) {
    int e = blockIdx.x * blockDim.x + threadIdx.x;
    if (e >= E) return;
    int r = rcv[e];
#pragma unroll
    for (int h = 0; h < HEADS; ++h) {
        float exr = __expf(rlbuf[(size_t)h * E + e] - maxr[h * N + r]);
        float ext = __expf(tlbuf[(size_t)h * E + e] - maxt[h * N + r]);
        rlbuf[(size_t)h * E + e] = exr;
        tlbuf[(size_t)h * E + e] = ext;
        atomicAdd(&denr[h * N + r], exr);
        atomicAdd(&dent[h * N + r], ext);
    }
}

// ---------------- 6. message pass + aggregation ----------------
// 64 threads per edge (one per feature), heads summed before the atomic
__global__ void edge_msg_kernel(const int* __restrict__ snd, const int* __restrict__ rcv,
                                const float* __restrict__ elen,
                                const float* __restrict__ exr, const float* __restrict__ ext,
                                const float* __restrict__ denr, const float* __restrict__ dent,
                                const float* __restrict__ mix_bias, const float* __restrict__ mix_scale,
                                const _Float16* __restrict__ rt_tab, const _Float16* __restrict__ tt_tab,
                                float* __restrict__ agg, int N, int E) {
    int tid = threadIdx.x;
    int e = blockIdx.x * 4 + (tid >> 6);
    int f = tid & 63;
    if (e >= E) return;
    int s = snd[e], r = rcv[e];
    float len = elen[e];
    float val = 0.f;
#pragma unroll
    for (int h = 0; h < HEADS; ++h) {
        float mix = 1.f / (1.f + __expf(-(mix_bias[h] + mix_scale[h] * len)));
        float wr = mix * exr[(size_t)h * E + e] / denr[h * N + r];
        float wt = (1.f - mix) * ext[(size_t)h * E + e] / dent[h * N + r];
        size_t bs = ((size_t)h * N + s) * FDIM + f;
        size_t br = ((size_t)h * N + r) * FDIM + f;
        val += wr * ((float)rt_tab[bs] - (float)rt_tab[br])
             + wt * ((float)tt_tab[bs] - (float)tt_tab[br]);
    }
    atomicAdd(&agg[(size_t)r * FDIM + f], val);
}

// ---------------- 7. final: xn + (agg/H)@Wout*layer_scale via WMMA ----------------
__global__ void final_kernel(const float* __restrict__ agg, const float* __restrict__ xn,
                             const _Float16* __restrict__ woT, const float* __restrict__ lscale,
                             float* __restrict__ out, int N, int NT) {
    int wid  = (blockIdx.x * blockDim.x + threadIdx.x) >> 5;
    int lane = threadIdx.x & 31;
    if (wid >= NT) return;
    int tile = wid;
    int lrow = lane & 15;
    int lhi  = (lane >> 4) & 1;
    int row  = tile * 16 + lrow;
    if (row >= N) row = N - 1;

    v16h a0, a1;
#pragma unroll
    for (int i = 0; i < 8; ++i) {
        int k0 = ((i < 4) ? 0 : 16) + lhi * 8 + (i & 3) * 2;
        const float* ar = agg + (size_t)row * FDIM;
        a0[2 * i]     = (_Float16)(n2nf(ar[k0])      * 0.25f);
        a0[2 * i + 1] = (_Float16)(n2nf(ar[k0 + 1])  * 0.25f);
        a1[2 * i]     = (_Float16)(n2nf(ar[32 + k0]) * 0.25f);
        a1[2 * i + 1] = (_Float16)(n2nf(ar[33 + k0]) * 0.25f);
    }
    bool full = (tile * 16 + 16 <= N);
#pragma unroll
    for (int nt = 0; nt < 4; ++nt) {
        int col = nt * 16 + lrow;
        const h2* wr = (const h2*)(woT + (size_t)col * FDIM);
        v16h b0, b1;
#pragma unroll
        for (int i = 0; i < 8; ++i) {
            int k0 = ((i < 4) ? 0 : 16) + lhi * 8 + (i & 3) * 2;
            h2 p0 = wr[k0 >> 1];
            h2 p1 = wr[(32 + k0) >> 1];
            b0[2 * i] = p0[0]; b0[2 * i + 1] = p0[1];
            b1[2 * i] = p1[0]; b1[2 * i + 1] = p1[1];
        }
        v8f c = {};
        c = __builtin_amdgcn_wmma_f32_16x16x32_f16(false, a0, false, b0, (short)0, c, false, false);
        c = __builtin_amdgcn_wmma_f32_16x16x32_f16(false, a1, false, b1, (short)0, c, false, false);
        float ls = lscale[col];
        if (full) {
#pragma unroll
            for (int g2 = 0; g2 < 8; ++g2) {
                int r2 = tile * 16 + g2 + lhi * 8;
                out[(size_t)r2 * FDIM + col] = xn[(size_t)r2 * FDIM + col] + c[g2] * ls;
            }
        } else {
#pragma unroll
            for (int g2 = 0; g2 < 8; ++g2) {
                int r2 = tile * 16 + g2 + lhi * 8;
                if (r2 < N)
                    out[(size_t)r2 * FDIM + col] = xn[(size_t)r2 * FDIM + col] + c[g2] * ls;
            }
        }
    }
}

// ---------------- launch ----------------
extern "C" void kernel_launch(void* const* d_in, const int* in_sizes, int n_in,
                              void* d_out, int out_size, void* d_ws, size_t ws_size,
                              hipStream_t stream) {
    const float* x           = (const float*)d_in[0];
    const int*   snd         = (const int*)d_in[1];
    const int*   rcv         = (const int*)d_in[2];
    /* d_in[3] edge_vec unused */
    const float* elen        = (const float*)d_in[4];
    const float* We          = (const float*)d_in[5];
    const float* Wr          = (const float*)d_in[6];
    const float* Wt          = (const float*)d_in[7];
    const float* Wout        = (const float*)d_in[8];
    const float* rs          = (const float*)d_in[9];
    const float* ts          = (const float*)d_in[10];
    const float* log_scale   = (const float*)d_in[11];
    const float* temp_bias   = (const float*)d_in[12];
    const float* temp_weight = (const float*)d_in[13];
    const float* mix_bias    = (const float*)d_in[14];
    const float* mix_scale   = (const float*)d_in[15];
    const float* Wd1         = (const float*)d_in[16];
    const float* bd1         = (const float*)d_in[17];
    const float* Wd2         = (const float*)d_in[18];
    const float* bd2         = (const float*)d_in[19];
    const float* Wt1         = (const float*)d_in[20];
    const float* bt1         = (const float*)d_in[21];
    const float* Wt2         = (const float*)d_in[22];
    const float* bt2         = (const float*)d_in[23];
    const float* ln_g        = (const float*)d_in[24];
    const float* ln_b        = (const float*)d_in[25];
    const float* lscale      = (const float*)d_in[26];
    float*       outp        = (float*)d_out;

    const int N = in_sizes[0] / FDIM;
    const int E = in_sizes[1];
    const int NT = (N + 15) / 16;

    // bump allocator over workspace (256B aligned)
    char* p = (char*)d_ws;
    auto alloc = [&](size_t bytes) -> char* {
        char* q = p;
        p += (bytes + 255) & ~(size_t)255;
        return q;
    };
    float*    xn    = (float*)   alloc((size_t)N * FDIM * 4);
    _Float16* xnb   = (_Float16*)alloc((size_t)N * FDIM * 2);
    _Float16* wT    = (_Float16*)alloc((size_t)13 * FDIM * FDIM * 2);
    _Float16* et    = (_Float16*)alloc((size_t)HEADS * N * FDIM * 2);
    _Float16* rtb   = (_Float16*)alloc((size_t)HEADS * N * FDIM * 2);
    _Float16* ttb   = (_Float16*)alloc((size_t)HEADS * N * FDIM * 2);
    float*    se    = (float*)   alloc((size_t)HEADS * N * 4);
    float*    st    = (float*)   alloc((size_t)HEADS * N * 4);
    float*    hdec  = (float*)   alloc((size_t)HEADS * N * 4);
    float*    htmp  = (float*)   alloc((size_t)HEADS * N * 4);
    float*    rlbuf = (float*)   alloc((size_t)HEADS * E * 4);
    float*    tlbuf = (float*)   alloc((size_t)HEADS * E * 4);
    // zero-init block (contiguous): agg, denr, dent
    char* zbeg = p;
    float* agg  = (float*)alloc((size_t)N * FDIM * 4);
    float* denr = (float*)alloc((size_t)HEADS * N * 4);
    float* dent = (float*)alloc((size_t)HEADS * N * 4);
    size_t zbytes = (size_t)(p - zbeg);
    // 0xFF-init block (contiguous): maxr, maxt  (works with atomicMaxFloat trick)
    char* fbeg = p;
    float* maxr = (float*)alloc((size_t)HEADS * N * 4);
    float* maxt = (float*)alloc((size_t)HEADS * N * 4);
    size_t fbytes = (size_t)(p - fbeg);

    (void)hipMemsetAsync(zbeg, 0x00, zbytes, stream);
    (void)hipMemsetAsync(fbeg, 0xFF, fbytes, stream);

    // 0. transpose + convert all weight matrices to f16 column-major
    prep_w_kernel<<<(13 * FDIM * FDIM + 255) / 256, 256, 0, stream>>>(We, Wr, Wt, Wout, wT);

    // 1. layernorm (one wave per node row)
    ln_kernel<<<(N + 7) / 8, 256, 0, stream>>>(x, ln_g, ln_b, xn, xnb, N);

    // 2. projections: 3 matrices x H heads x NT row-tiles, one wave each
    {
        int waves = 3 * HEADS * NT;
        proj_kernel<<<(waves + 7) / 8, 256, 0, stream>>>(xnb, wT, et, rtb, ttb, N, NT);
    }

    // 3. per-(head,node) scores + MLPs
    nodehead_kernel<<<(HEADS * N + 255) / 256, 256, 0, stream>>>(
        et, rs, ts, Wd1, bd1, Wd2, bd2, Wt1, bt1, Wt2, bt2, se, st, hdec, htmp, N);

    // 4. edge logits + segment max
    edge_logits_kernel<<<(E + 255) / 256, 256, 0, stream>>>(
        snd, rcv, elen, se, st, hdec, htmp, log_scale, temp_bias, temp_weight,
        rlbuf, tlbuf, maxr, maxt, N, E);

    // 5. exp + segment sum
    edge_exp_kernel<<<(E + 255) / 256, 256, 0, stream>>>(
        rcv, rlbuf, tlbuf, maxr, maxt, denr, dent, N, E);

    // 6. message + aggregation (4 edges per 256-thread block)
    edge_msg_kernel<<<(E + 3) / 4, 256, 0, stream>>>(
        snd, rcv, elen, rlbuf, tlbuf, denr, dent, mix_bias, mix_scale, rtb, ttb, agg, N, E);

    // 7. final GEMM + epilogue
    final_kernel<<<(NT + 7) / 8, 256, 0, stream>>>(agg, xn, wT + (size_t)12 * FDIM * FDIM,
                                                   lscale, outp, N, NT);
}
